// GCNNet_66391604462260
// MI455X (gfx1250) — matrix operand order repaired and checked by reading, not compiled, not run on hardware
//
#include <hip/hip_runtime.h>
#include <hip/hip_bf16.h>

// ---------------------------------------------------------------------------
// Types
// ---------------------------------------------------------------------------
typedef __bf16 bf16;
typedef __attribute__((ext_vector_type(16))) __bf16 v16bf;
typedef __attribute__((ext_vector_type(8)))  __bf16 v8bf;
typedef __attribute__((ext_vector_type(8)))  float  v8f;

#define N_NODES  20480
#define N_EDGES  327680
#define N_GRAPHS 256

// ---------------------------------------------------------------------------
// Helpers
// ---------------------------------------------------------------------------
__device__ __forceinline__ bf16 f2bf(float f) {
    unsigned u = __float_as_uint(f);
    unsigned r = (u + 0x7FFFu + ((u >> 16) & 1u)) >> 16;   // RNE
    unsigned short s = (unsigned short)r;
    bf16 h;
    __builtin_memcpy(&h, &s, 2);
    return h;
}

__device__ __forceinline__ bf16 bits2bf(unsigned short s) {
    bf16 h;
    __builtin_memcpy(&h, &s, 2);
    return h;
}

// ---------------------------------------------------------------------------
// Elementwise / setup kernels
// ---------------------------------------------------------------------------
__global__ void k_zero_f32(float* __restrict__ p, long long n) {
    long long i = (long long)blockIdx.x * blockDim.x + threadIdx.x;
    if (i < n) p[i] = 0.0f;
}

__global__ void k_f32_to_bf16(const float* __restrict__ src, bf16* __restrict__ dst,
                              long long n) {
    long long i = (long long)blockIdx.x * blockDim.x + threadIdx.x;
    if (i < n) dst[i] = f2bf(src[i]);
}

__global__ void k_degree(const int* __restrict__ dst_idx, int* __restrict__ deg, int E) {
    int i = blockIdx.x * blockDim.x + threadIdx.x;
    if (i < E) atomicAdd(&deg[dst_idx[i]], 1);
}

__global__ void k_counts(const int* __restrict__ batch, int* __restrict__ cnt, int n) {
    int i = blockIdx.x * blockDim.x + threadIdx.x;
    if (i < n) atomicAdd(&cnt[batch[i]], 1);
}

__global__ void k_invdeg(const int* __restrict__ deg, float* __restrict__ dis,
                         float* __restrict__ inv, int n) {
    int i = blockIdx.x * blockDim.x + threadIdx.x;
    if (i < n) {
        float d = (float)deg[i] + 1.0f;   // + self loop
        dis[i] = rsqrtf(d);
        inv[i] = 1.0f / d;
    }
}

// ---------------------------------------------------------------------------
// WMMA bf16 GEMM: C[M,N] = act(A[M,K] @ B[K,N] + bias)
// Block = 256 threads = 8 waves. Block tile 128 (M) x 64 (N), K step 32.
// Each wave owns 16 M-rows and 4 accumulators (4 x 16 N-cols): the A fragment
// is reused across 4 WMMAs per K-step.
// B tiles are staged through double-buffered LDS, pre-swizzled to the
// ISA 7.12.2 B-fragment layout; the next tile's global load (one b128/thread)
// and the next A fragments are prefetched into registers while the current
// 4 WMMAs issue, so the loadcnt wait lands after the compute.
// ---------------------------------------------------------------------------
__global__ void k_gemm_bf16_wmma(const bf16* __restrict__ A, const bf16* __restrict__ B,
                                 const float* __restrict__ bias,
                                 float* __restrict__ Cf, bf16* __restrict__ Cb,
                                 int M, int N, int K, int ldc, int relu) {
    __shared__ bf16 Bs[2][2048];   // [buf][tile(4) * 512 + lane(32)*16 + e(16)]

    const int t    = threadIdx.x;
    const int lane = t & 31;
    const int wave = t >> 5;
    const int half = lane >> 4;
    const int n    = lane & 15;
    const int n0   = blockIdx.y * 64;
    const int m0w  = blockIdx.x * 128 + wave * 16;

    const bf16* Arow = A + (size_t)(m0w + n) * (size_t)K + half * 8;

    // B staging geometry: 32 (K) x 64 (N) tile = 1024 dwords; thread t loads
    // dwords 4t..4t+3 (one b128), all inside row Kr with col pairs c2b..c2b+6.
    const int Kr  = t >> 3;            // 0..31
    const int c2b = (t & 7) * 8;       // 0,8,...,56
    const int l0  = ((Kr >> 3) & 1) * 16;
    const int e   = (Kr & 7) | ((Kr >> 4) << 3);

    v8f acc[4];
    #pragma unroll
    for (int j = 0; j < 4; ++j) acc[j] = (v8f){0.f,0.f,0.f,0.f,0.f,0.f,0.f,0.f};

    // ---- prologue: prefetch k0 = 0 ----
    v8bf a0c = *(const v8bf*)(Arow);
    v8bf a1c = *(const v8bf*)(Arow + 16);
    uint4 bqc = *(const uint4*)(B + (size_t)Kr * (size_t)N + n0 + c2b);

    int buf = 0;
    for (int k0 = 0; k0 < K; k0 += 32) {
        // ---- commit staged B tile to LDS (fragment-swizzled) ----
        {
            unsigned uu[4] = {bqc.x, bqc.y, bqc.z, bqc.w};
            #pragma unroll
            for (int i = 0; i < 4; ++i) {
                int col  = c2b + 2 * i;
                int tile = col >> 4;
                int nn   = col & 15;
                int idx  = tile * 512 + (l0 + nn) * 16 + e;
                Bs[buf][idx]      = bits2bf((unsigned short)uu[i]);
                Bs[buf][idx + 16] = bits2bf((unsigned short)(uu[i] >> 16));
            }
        }
        __syncthreads();

        // ---- prefetch next K-step (global loads overlap current WMMAs) ----
        const bool more = (k0 + 32) < K;
        v8bf a0n = a0c, a1n = a1c;
        uint4 bqn = bqc;
        if (more) {
            const bf16* an = Arow + (k0 + 32);
            a0n = *(const v8bf*)(an);
            a1n = *(const v8bf*)(an + 16);
            bqn = *(const uint4*)(B + (size_t)(k0 + 32 + Kr) * (size_t)N + n0 + c2b);
        }

        // ---- compute: 4 WMMAs reusing one A fragment ----
        v16bf av;
        #pragma unroll
        for (int i = 0; i < 8; ++i) { av[i] = a0c[i]; av[i + 8] = a1c[i]; }

        #pragma unroll
        for (int j = 0; j < 4; ++j) {
            const bf16* bp = &Bs[buf][j * 512 + lane * 16];
            v8bf b0 = *(const v8bf*)(bp);
            v8bf b1 = *(const v8bf*)(bp + 8);
            v16bf bv;
            #pragma unroll
            for (int i = 0; i < 8; ++i) { bv[i] = b0[i]; bv[i + 8] = b1[i]; }
            acc[j] = __builtin_amdgcn_wmma_f32_16x16x32_bf16(
                false, av, false, bv, (short)0, acc[j], false, false);
        }

        a0c = a0n; a1c = a1n; bqc = bqn;
        buf ^= 1;
    }

    // ---- store: D element r -> row = r + 8*half, col = lane&15 ----
    #pragma unroll
    for (int j = 0; j < 4; ++j) {
        #pragma unroll
        for (int r = 0; r < 8; ++r) {
            int row = m0w + r + half * 8;
            int col = n0 + j * 16 + n;
            if (row < M) {
                float v = acc[j][r];
                if (bias) v += bias[col];
                if (relu) v = fmaxf(v, 0.0f);
                size_t off = (size_t)row * (size_t)ldc + col;
                Cf[off] = v;
                if (Cb) Cb[off] = f2bf(v);
            }
        }
    }
}

// ---------------------------------------------------------------------------
// Edge scatter: agg[dst] += H[src] * dis[src]*dis[dst]   (4 floats / thread)
// ---------------------------------------------------------------------------
__global__ void k_scatter(const float* __restrict__ H, const int* __restrict__ src,
                          const int* __restrict__ dst, const float* __restrict__ dis,
                          float* __restrict__ agg, int E, int D) {
    long long idx = (long long)blockIdx.x * blockDim.x + threadIdx.x;
    int chunks = D >> 2;
    long long total = (long long)E * chunks;
    if (idx >= total) return;
    int e  = (int)(idx / chunks);
    int c  = ((int)(idx % chunks)) << 2;
    int s  = src[e], d = dst[e];
    float nm = dis[s] * dis[d];
    const float4 v = *(const float4*)(H + (size_t)s * D + c);
    float* out = agg + (size_t)d * D + c;
    atomicAdd(out + 0, v.x * nm);
    atomicAdd(out + 1, v.y * nm);
    atomicAdd(out + 2, v.z * nm);
    atomicAdd(out + 3, v.w * nm);
}

// ---------------------------------------------------------------------------
// Finalize GCN layer: h = relu(agg + H/deg + b), write f32 (in place) + bf16
// ---------------------------------------------------------------------------
__global__ void k_finalize(float* __restrict__ agg, const float* __restrict__ H,
                           const float* __restrict__ inv, const float* __restrict__ bias,
                           bf16* __restrict__ hb, int n_nodes, int D) {
    long long idx = (long long)blockIdx.x * blockDim.x + threadIdx.x;
    long long total = (long long)n_nodes * D;
    if (idx >= total) return;
    int i = (int)(idx / D);
    int d = (int)(idx % D);
    float v = agg[idx] + H[idx] * inv[i] + bias[d];
    v = fmaxf(v, 0.0f);
    agg[idx] = v;
    hb[idx]  = f2bf(v);
}

// ---------------------------------------------------------------------------
// Pooling: per-graph sum + max (h >= 0 post-ReLU, so uint order == float order)
// ---------------------------------------------------------------------------
__global__ void k_pool(const float* __restrict__ h, const int* __restrict__ batch,
                       float* __restrict__ gap, unsigned int* __restrict__ gmp,
                       int n_nodes, int D) {
    long long idx = (long long)blockIdx.x * blockDim.x + threadIdx.x;
    long long total = (long long)n_nodes * D;
    if (idx >= total) return;
    int i = (int)(idx / D);
    int d = (int)(idx % D);
    int g = batch[i];
    float v = h[idx];
    atomicAdd(&gap[(size_t)g * D + d], v);
    atomicMax(&gmp[(size_t)g * D + d], __float_as_uint(v));
}

__global__ void k_poolfin(const float* __restrict__ gap, const unsigned int* __restrict__ gmp,
                          const int* __restrict__ cnt, float* __restrict__ gfull,
                          bf16* __restrict__ gfullb, int D) {
    int idx = blockIdx.x * blockDim.x + threadIdx.x;  // 256*512
    if (idx >= N_GRAPHS * D) return;
    int g = idx / D;
    int d = idx % D;
    float c  = (float)cnt[g];
    float ga = gap[idx] / fmaxf(c, 1.0f);
    float gm = (c > 0.0f) ? __uint_as_float(gmp[idx]) : 0.0f;
    size_t o0 = (size_t)g * (2 * D) + d;
    size_t o1 = o0 + D;
    gfull[o0] = ga;  gfull[o1] = gm;
    gfullb[o0] = f2bf(ga);  gfullb[o1] = f2bf(gm);
}

// ---------------------------------------------------------------------------
// Final 512 -> 1 layer (tiny): one thread per graph
// ---------------------------------------------------------------------------
__global__ void k_dot_out(const float* __restrict__ z, const float* __restrict__ Wo,
                          const float* __restrict__ bo, float* __restrict__ out) {
    int g = blockIdx.x * blockDim.x + threadIdx.x;
    if (g >= N_GRAPHS) return;
    const float* row = z + (size_t)g * 512;
    float s = 0.0f;
    #pragma unroll 4
    for (int k = 0; k < 512; k += 4) {
        float4 a = *(const float4*)(row + k);
        float4 b = *(const float4*)(Wo + k);
        s += a.x * b.x + a.y * b.y + a.z * b.z + a.w * b.w;
    }
    out[g] = s + bo[0];
}

// ---------------------------------------------------------------------------
// Launcher
// ---------------------------------------------------------------------------
extern "C" void kernel_launch(void* const* d_in, const int* in_sizes, int n_in,
                              void* d_out, int out_size, void* d_ws, size_t ws_size,
                              hipStream_t stream) {
    (void)in_sizes; (void)n_in; (void)out_size; (void)ws_size;

    const float* x    = (const float*)d_in[0];
    const int*   eix  = (const int*)  d_in[1];
    const int*   batch= (const int*)  d_in[3];
    const float* solv = (const float*)d_in[4];
    const float* W1 = (const float*)d_in[5];  const float* b1 = (const float*)d_in[6];
    const float* W2 = (const float*)d_in[7];  const float* b2 = (const float*)d_in[8];
    const float* W3 = (const float*)d_in[9];  const float* b3 = (const float*)d_in[10];
    const float* Wg1= (const float*)d_in[11]; const float* bg1= (const float*)d_in[12];
    const float* Wg2= (const float*)d_in[13]; const float* bg2= (const float*)d_in[14];
    const float* Ws1= (const float*)d_in[15]; const float* bs1= (const float*)d_in[16];
    const float* Ws2= (const float*)d_in[17]; const float* bs2= (const float*)d_in[18];
    const float* Wf1= (const float*)d_in[19]; const float* bf1= (const float*)d_in[20];
    const float* Wf2= (const float*)d_in[21]; const float* bf2= (const float*)d_in[22];
    const float* Wo = (const float*)d_in[23]; const float* bo = (const float*)d_in[24];
    const int* src = eix;
    const int* dst = eix + N_EDGES;
    float* out = (float*)d_out;

    // ---- workspace layout (256B aligned) ----
    char* ws = (char*)d_ws;
    size_t off = 0;
    auto alloc = [&](size_t bytes) -> void* {
        off = (off + 255) & ~(size_t)255;
        void* p = ws + off;
        off += bytes;
        return p;
    };
    bf16*  xb    = (bf16*) alloc((size_t)N_NODES * 128 * 2);
    bf16*  wbuf  = (bf16*) alloc((size_t)1024 * 1024 * 2);
    float* Hf    = (float*)alloc((size_t)N_NODES * 512 * 4);
    float* Ag    = (float*)alloc((size_t)N_NODES * 512 * 4);
    bf16*  hb    = (bf16*) alloc((size_t)N_NODES * 512 * 2);
    int*   deg   = (int*)  alloc((size_t)N_NODES * 4);
    int*   cnt   = (int*)  alloc((size_t)N_GRAPHS * 4);
    float* dis   = (float*)alloc((size_t)N_NODES * 4);
    float* inv   = (float*)alloc((size_t)N_NODES * 4);
    float* gap   = (float*)alloc((size_t)N_GRAPHS * 512 * 4);
    unsigned int* gmp = (unsigned int*)alloc((size_t)N_GRAPHS * 512 * 4);
    float* gfull = (float*)alloc((size_t)N_GRAPHS * 1024 * 4);
    bf16*  gfullb= (bf16*) alloc((size_t)N_GRAPHS * 1024 * 2);
    float* scr   = (float*)alloc((size_t)N_GRAPHS * 1024 * 4);
    bf16*  g1b   = (bf16*) alloc((size_t)N_GRAPHS * 1024 * 2);
    bf16*  zb    = (bf16*) alloc((size_t)N_GRAPHS * 256 * 2);
    bf16*  s1b   = (bf16*) alloc((size_t)N_GRAPHS * 256 * 2);
    bf16*  z1b   = (bf16*) alloc((size_t)N_GRAPHS * 1024 * 2);
    float* z2f   = (float*)alloc((size_t)N_GRAPHS * 512 * 4);
    bf16*  solvb = (bf16*) alloc((size_t)N_GRAPHS * 512 * 2);

    auto blocks = [](long long n) { return (unsigned)((n + 255) / 256); };
    auto zero_f = [&](void* p, long long n) {
        k_zero_f32<<<blocks(n), 256, 0, stream>>>((float*)p, n);
    };
    auto to_bf = [&](const float* s, bf16* d, long long n) {
        k_f32_to_bf16<<<blocks(n), 256, 0, stream>>>(s, d, n);
    };
    auto gemm = [&](const bf16* A, const bf16* B, const float* bias, float* Cf,
                    bf16* Cb, int M, int N, int K, int ldc, int relu) {
        dim3 grid((M + 127) / 128, N / 64);
        k_gemm_bf16_wmma<<<grid, 256, 0, stream>>>(A, B, bias, Cf, Cb, M, N, K, ldc, relu);
    };

    // ---- degrees, graph counts, normalizers ----
    zero_f(deg, N_NODES);
    zero_f(cnt, N_GRAPHS);
    k_degree<<<blocks(N_EDGES), 256, 0, stream>>>(dst, deg, N_EDGES);
    k_counts<<<blocks(N_NODES), 256, 0, stream>>>(batch, cnt, N_NODES);
    k_invdeg<<<blocks(N_NODES), 256, 0, stream>>>(deg, dis, inv, N_NODES);

    // ---- input conversions ----
    to_bf(x,    xb,    (long long)N_NODES * 128);
    to_bf(solv, solvb, (long long)N_GRAPHS * 512);

    // ---- GCN layer 1 (128 -> 128) ----
    to_bf(W1, wbuf, 128 * 128);
    gemm(xb, wbuf, nullptr, Hf, nullptr, N_NODES, 128, 128, 128, 0);
    zero_f(Ag, (long long)N_NODES * 128);
    k_scatter<<<blocks((long long)N_EDGES * 32), 256, 0, stream>>>(Hf, src, dst, dis, Ag, N_EDGES, 128);
    k_finalize<<<blocks((long long)N_NODES * 128), 256, 0, stream>>>(Ag, Hf, inv, b1, hb, N_NODES, 128);

    // ---- GCN layer 2 (128 -> 256) ----
    to_bf(W2, wbuf, 128 * 256);
    gemm(hb, wbuf, nullptr, Hf, nullptr, N_NODES, 256, 128, 256, 0);
    zero_f(Ag, (long long)N_NODES * 256);
    k_scatter<<<blocks((long long)N_EDGES * 64), 256, 0, stream>>>(Hf, src, dst, dis, Ag, N_EDGES, 256);
    k_finalize<<<blocks((long long)N_NODES * 256), 256, 0, stream>>>(Ag, Hf, inv, b2, hb, N_NODES, 256);

    // ---- GCN layer 3 (256 -> 512) ----
    to_bf(W3, wbuf, 256 * 512);
    gemm(hb, wbuf, nullptr, Hf, nullptr, N_NODES, 512, 256, 512, 0);
    zero_f(Ag, (long long)N_NODES * 512);
    k_scatter<<<blocks((long long)N_EDGES * 128), 256, 0, stream>>>(Hf, src, dst, dis, Ag, N_EDGES, 512);
    k_finalize<<<blocks((long long)N_NODES * 512), 256, 0, stream>>>(Ag, Hf, inv, b3, hb, N_NODES, 512);
    // Ag now holds h3 (f32, [N,512])

    // ---- pooling: gap | gmp -> gfull [256,1024] ----
    zero_f(gap, (long long)N_GRAPHS * 512);
    zero_f(gmp, (long long)N_GRAPHS * 512);
    k_pool<<<blocks((long long)N_NODES * 512), 256, 0, stream>>>(Ag, batch, gap, gmp, N_NODES, 512);
    k_poolfin<<<blocks((long long)N_GRAPHS * 512), 256, 0, stream>>>(gap, gmp, cnt, gfull, gfullb, 512);

    // ---- MLP head ----
    to_bf(Wg1, wbuf, 1024 * 1024);
    gemm(gfullb, wbuf, bg1, scr, g1b, N_GRAPHS, 1024, 1024, 1024, 1);
    to_bf(Wg2, wbuf, 1024 * 128);
    gemm(g1b, wbuf, bg2, scr, zb, N_GRAPHS, 128, 1024, 256, 1);        // z[:, 0:128]
    to_bf(Ws1, wbuf, 512 * 256);
    gemm(solvb, wbuf, bs1, scr, s1b, N_GRAPHS, 256, 512, 256, 1);
    to_bf(Ws2, wbuf, 256 * 128);
    gemm(s1b, wbuf, bs2, scr, zb + 128, N_GRAPHS, 128, 256, 256, 1);   // z[:, 128:256]
    to_bf(Wf1, wbuf, 256 * 1024);
    gemm(zb, wbuf, bf1, scr, z1b, N_GRAPHS, 1024, 256, 1024, 1);
    to_bf(Wf2, wbuf, 1024 * 512);
    gemm(z1b, wbuf, bf2, z2f, nullptr, N_GRAPHS, 512, 1024, 512, 1);

    k_dot_out<<<1, 256, 0, stream>>>(z2f, Wo, bo, out);
}
